// CalAttention_18296560681459
// MI455X (gfx1250) — matrix operand order repaired and checked by reading, not compiled, not run on hardware
//
#include <hip/hip_runtime.h>
#include <stdint.h>

typedef _Float16 f16;
typedef __attribute__((ext_vector_type(8)))  f16   v8h;
typedef __attribute__((ext_vector_type(16))) f16   v16h;
typedef __attribute__((ext_vector_type(8)))  float v8f;

union AF { v16h v; struct { v8h lo, hi; } p; };

constexpr int Bc = 2, Hc = 8, Sc = 4096, Dc = 64;
// fold 1/sqrt(64) AND log2(e) into the Q pre-scale: softmax computed in base-2 domain,
// 2^(s - m) == e^(score - m_e); exponential is then one raw v_exp_f32.
constexpr float QSCALE = 0.125f * 1.44269504088896340736f;
constexpr int KT  = 64;                // keys per tile
constexpr int NT  = Sc / KT;           // 64 tiles
constexpr int KPAD = 72;               // row strides (halfs): 144B rows, 16B aligned
constexpr int VPAD = 72;
constexpr int PPAD = 72;

__device__ __forceinline__ float fast_exp2(float x) {
    return __builtin_amdgcn_exp2f(x);  // single v_exp_f32, no libm wrapper
}

__global__ __launch_bounds__(256)
void flash_attn_f16wmma_kernel(const float* __restrict__ Q,
                               const float* __restrict__ K,
                               const float* __restrict__ V,
                               const uint8_t* __restrict__ M,
                               float* __restrict__ O)
{
    __shared__ __align__(16) f16 sK[2][KT][KPAD];  // [buf][key][d]
    __shared__ __align__(16) f16 sV[2][Dc][VPAD];  // [buf][d][key]  (transposed)
    __shared__ __align__(16) f16 sP[8][16][PPAD];  // per-wave P^T as [q][k]

    const int tid  = threadIdx.x;
    const int wave = tid >> 5;
    const int lane = tid & 31;
    const int l16  = lane & 15;
    const int lh   = lane >> 4;

    const int bh    = blockIdx.y;
    const int b     = bh >> 3;                 // H == 8
    const int qbase = blockIdx.x * 128 + wave * 16;
    const int qrow  = qbase + l16;             // this lane's query row (fixed)

    const float*   Qb   = Q + (size_t)bh * Sc * Dc;
    const float*   Kb   = K + (size_t)bh * Sc * Dc;
    const float*   Vb   = V + (size_t)bh * Sc * Dc;
    const uint8_t* Mrow = M + (size_t)b * Sc * Sc + (size_t)qrow * Sc;

    // ---- Q as B-matrix fragments: lane = column q=l16, K chunks = d[dc*32+lh*8, +8) and d[dc*32+16+lh*8, +8) ----
    AF qf[2];
    #pragma unroll
    for (int dc = 0; dc < 2; ++dc) {
        const float* qp = Qb + (size_t)qrow * Dc + dc * 32 + lh * 8;
        v8h lo, hi;
        #pragma unroll
        for (int i = 0; i < 8; ++i) {
            lo[i] = (f16)(qp[i]     * QSCALE);
            hi[i] = (f16)(qp[16 + i] * QSCALE);
        }
        qf[dc].p.lo = lo; qf[dc].p.hi = hi;
    }

    float m_s = -INFINITY, l_s = 0.0f;         // scalar softmax state per lane (base-2)
    v8f Of[4] = { {}, {}, {}, {} };            // O^T accumulators: M=d, N=q

    // ---- cooperative staging: each thread owns key rows (r2, r2+1) x 8 d-values ----
    const int r2 = (tid >> 3) * 2;             // 0,2,..,62
    const int sd = (tid & 7) * 8;              // d offset
    float4 k0a,k0b,k1a,k1b, v0a,v0b,v1a,v1b;

    auto stage_load = [&](int t) {
        const float* kp = Kb + (size_t)(t * KT + r2) * Dc + sd;
        const float* vp = Vb + (size_t)(t * KT + r2) * Dc + sd;
        k0a = *(const float4*)kp;            k0b = *(const float4*)(kp + 4);
        k1a = *(const float4*)(kp + Dc);     k1b = *(const float4*)(kp + Dc + 4);
        v0a = *(const float4*)vp;            v0b = *(const float4*)(vp + 4);
        v1a = *(const float4*)(vp + Dc);     v1b = *(const float4*)(vp + Dc + 4);
    };
    auto stage_store = [&](int buf) {
        v8h kh0, kh1;
        kh0[0]=(f16)k0a.x; kh0[1]=(f16)k0a.y; kh0[2]=(f16)k0a.z; kh0[3]=(f16)k0a.w;
        kh0[4]=(f16)k0b.x; kh0[5]=(f16)k0b.y; kh0[6]=(f16)k0b.z; kh0[7]=(f16)k0b.w;
        kh1[0]=(f16)k1a.x; kh1[1]=(f16)k1a.y; kh1[2]=(f16)k1a.z; kh1[3]=(f16)k1a.w;
        kh1[4]=(f16)k1b.x; kh1[5]=(f16)k1b.y; kh1[6]=(f16)k1b.z; kh1[7]=(f16)k1b.w;
        *reinterpret_cast<v8h*>(&sK[buf][r2][sd])     = kh0;
        *reinterpret_cast<v8h*>(&sK[buf][r2 + 1][sd]) = kh1;
        // transposed V: adjacent key cols packed as one dword per d
        const float va[8] = {v0a.x,v0a.y,v0a.z,v0a.w,v0b.x,v0b.y,v0b.z,v0b.w};
        const float vb[8] = {v1a.x,v1a.y,v1a.z,v1a.w,v1b.x,v1b.y,v1b.z,v1b.w};
        #pragma unroll
        for (int i = 0; i < 8; ++i) {
            union { f16 h[2]; uint32_t u; } pk;
            pk.h[0] = (f16)va[i]; pk.h[1] = (f16)vb[i];
            *reinterpret_cast<uint32_t*>(&sV[buf][sd + i][r2]) = pk.u;
        }
    };

    stage_load(0);
    stage_store(0);
    __syncthreads();

    for (int t = 0; t < NT; ++t) {
        const int buf   = t & 1;
        const int kbase = t * KT;
        if (t + 1 < NT) stage_load(t + 1);   // overlap global latency with compute

        // ---- S^T = K Q^T : four subtiles of 16 kcols x 16 qrows ----
        v8f s[4];
        #pragma unroll
        for (int st = 0; st < 4; ++st) {
            v8f c = {};
            #pragma unroll
            for (int dc = 0; dc < 2; ++dc) {
                AF kf;  // A-layout: lane row = kcol, half-dependent d chunks
                const f16* p = &sK[buf][st * 16 + l16][dc * 32 + lh * 8];
                kf.p.lo = *(const v8h*)p;
                kf.p.hi = *(const v8h*)(p + 16);
                c = __builtin_amdgcn_wmma_f32_16x16x32_f16(false, kf.v, false, qf[dc].v,
                                                           (short)0, c, false, false);
            }
            s[st] = c;
        }

        // ---- mask: one 8-byte load per subtile; cvt_f32_ubyte + fma(-1e9) ----
        #pragma unroll
        for (int st = 0; st < 4; ++st) {
            const uint2 mu = *(const uint2*)(Mrow + kbase + st * 16 + lh * 8);
            #pragma unroll
            for (int r = 0; r < 8; ++r) {
                const uint32_t w  = (r < 4) ? mu.x : mu.y;
                const float    mb = (float)((w >> (8 * (r & 3))) & 0xffu);
                s[st][r] = __builtin_fmaf(mb, -1e9f, s[st][r]);
            }
        }

        // ---- online softmax, base-2 (serial chains: low register pressure) ----
        float mt = s[0][0];
        #pragma unroll
        for (int st = 0; st < 4; ++st)
            #pragma unroll
            for (int r = (st == 0 ? 1 : 0); r < 8; ++r) mt = fmaxf(mt, s[st][r]);
        mt = fmaxf(mt, __shfl_xor(mt, 16, 32));

        const float mn    = fmaxf(m_s, mt);
        const float alpha = fast_exp2(m_s - mn);   // single v_exp_f32
        m_s = mn;

        float rs = 0.0f;
        #pragma unroll
        for (int st = 0; st < 4; ++st) {
            v8h ph;
            #pragma unroll
            for (int r = 0; r < 8; ++r) {
                const float p = fast_exp2(s[st][r] - mn);
                rs += p;
                ph[r] = (f16)p;
            }
            // P^T[q=l16][k = st*16 + lh*8 + r] : contiguous -> one b128 store
            *reinterpret_cast<v8h*>(&sP[wave][l16][st * 16 + lh * 8]) = ph;
        }
        rs += __shfl_xor(rs, 16, 32);
        l_s = l_s * alpha + rs;

        // ---- rescale O^T ----
        #pragma unroll
        for (int nb = 0; nb < 4; ++nb)
            #pragma unroll
            for (int r = 0; r < 8; ++r) Of[nb][r] *= alpha;

        // ---- O^T += V^T P^T  (K-dim = 64 keys = two 32-key chunks) ----
        AF pf[2];  // B = P^T: lane col q=l16, K chunks = k in [kc*32 + lh*8, +8) and [kc*32 + 16 + lh*8, +8)
        #pragma unroll
        for (int kc = 0; kc < 2; ++kc) {
            const f16* pp = &sP[wave][l16][kc * 32 + lh * 8];
            pf[kc].p.lo = *(const v8h*)pp;
            pf[kc].p.hi = *(const v8h*)(pp + 16);
        }
        #pragma unroll
        for (int nb = 0; nb < 4; ++nb) {
            #pragma unroll
            for (int kc = 0; kc < 2; ++kc) {
                AF vf;  // A = V^T: lane row d = nb*16+l16, half-dependent k chunks
                const f16* vp = &sV[buf][nb * 16 + l16][kc * 32 + lh * 8];
                vf.p.lo = *(const v8h*)vp;
                vf.p.hi = *(const v8h*)(vp + 16);
                Of[nb] = __builtin_amdgcn_wmma_f32_16x16x32_f16(false, vf.v, false, pf[kc].v,
                                                                (short)0, Of[nb], false, false);
            }
        }

        if (t + 1 < NT) {
            stage_store(buf ^ 1);   // readers of buf^1 finished before entering iter t
            __syncthreads();
        }
    }

    // ---- epilogue: out[q][d] = O^T[d][q] / l  (8 consecutive d per lane per nb) ----
    float* Orow = O + (size_t)bh * Sc * Dc + (size_t)qrow * Dc;
    const float inv = 1.0f / l_s;
    #pragma unroll
    for (int nb = 0; nb < 4; ++nb) {
        float4 a, bv;
        a.x  = Of[nb][0] * inv;  a.y  = Of[nb][1] * inv;
        a.z  = Of[nb][2] * inv;  a.w  = Of[nb][3] * inv;
        bv.x = Of[nb][4] * inv;  bv.y = Of[nb][5] * inv;
        bv.z = Of[nb][6] * inv;  bv.w = Of[nb][7] * inv;
        *(float4*)(Orow + nb * 16 + lh * 8)     = a;
        *(float4*)(Orow + nb * 16 + lh * 8 + 4) = bv;
    }
}

extern "C" void kernel_launch(void* const* d_in, const int* in_sizes, int n_in,
                              void* d_out, int out_size, void* d_ws, size_t ws_size,
                              hipStream_t stream) {
    const float*   q = (const float*)d_in[0];
    const float*   k = (const float*)d_in[1];
    const float*   v = (const float*)d_in[2];
    const uint8_t* m = (const uint8_t*)d_in[3];  // jax bool -> 1 byte per element
    float* out = (float*)d_out;
    (void)in_sizes; (void)n_in; (void)out_size; (void)d_ws; (void)ws_size;

    dim3 grid(Sc / 128, Bc * Hc);   // 32 q-blocks x 16 (b,h)
    dim3 block(256);                // 8 waves, 16 q-rows each
    flash_attn_f16wmma_kernel<<<grid, block, 0, stream>>>(q, k, v, m, out);
}
